// EarthAttention3D_21311627723325
// MI455X (gfx1250) — matrix operand order repaired and checked by reading, not compiled, not run on hardware
//
#include <hip/hip_runtime.h>
#include <hip/hip_bf16.h>

typedef __attribute__((ext_vector_type(16))) _Float16 v16h;
typedef __attribute__((ext_vector_type(8)))  float    v8f;

#define WMMA_F16(a, b, c) \
  __builtin_amdgcn_wmma_f32_16x16x32_f16(false, (a), false, (b), (short)0, (c), false, false)

// ---------- fragment loaders (wave32 CDNA5 WMMA layouts) ----------
// A 16x32 f16 (MxK): lanes 0-15 row M=lane, K=0..7 & 16..23 ; lanes 16-31 row M=lane-16, K=8..15 & 24..31
__device__ __forceinline__ v16h load_a_f16(const _Float16* base, int ld, int row0, int k0) {
  int lane = threadIdx.x & 31;
  int m  = row0 + (lane & 15);
  int kb = k0 + ((lane >> 4) << 3);
  const _Float16* p = base + (size_t)m * ld + kb;
  v16h a;
#pragma unroll
  for (int e = 0; e < 8; ++e) { a[e] = p[e]; a[e + 8] = p[16 + e]; }
  return a;
}

__device__ __forceinline__ v16h load_a_f32cvt(const float* base, int ld, int row0, int k0) {
  int lane = threadIdx.x & 31;
  int m  = row0 + (lane & 15);
  int kb = k0 + ((lane >> 4) << 3);
  const float* p = base + (size_t)m * ld + kb;
  v16h a;
#pragma unroll
  for (int e = 0; e < 8; ++e) { a[e] = (_Float16)p[e]; a[e + 8] = (_Float16)p[16 + e]; }
  return a;
}

// B 32x16 f16 (KxN): lanes 0-15 col N=lane hold K=0..7 & 16..23, lanes 16-31 col N=lane-16 hold K=8..15 & 24..31.
// Source layout must be [n][k] row-major (k contiguous), stride ld.
__device__ __forceinline__ v16h load_b_kcont(const _Float16* base, int ld, int col0, int k0) {
  int lane = threadIdx.x & 31;
  int n  = col0 + (lane & 15);
  int kb = k0 + ((lane >> 4) << 3);
  const _Float16* p = base + (size_t)n * ld + kb;
  v16h b;
#pragma unroll
  for (int e = 0; e < 8; ++e) { b[e] = p[e]; b[e + 8] = p[16 + e]; }
  return b;
}

// D 16x16 f32: vgpr e holds row M = e + 8*(lane/16), col N = lane%16
#define D_ROW(e)  ((e) + (((threadIdx.x & 31) >> 4) << 3))
#define D_COL()   ((threadIdx.x & 31) & 15)

// ---------- problem constants ----------
#define NTOK 144
#define DIM_ 384
#define HEADS_ 12
#define HD_ 32
#define BW_ 256
#define ROWS_ (BW_ * NTOK)           // 36864
#define QKVC (3 * DIM_)              // 1152

// ---------- kernel 1: transpose f32 weight -> f16 [col][k] ----------
__global__ void k_transpose_f16(const float* __restrict__ src, _Float16* __restrict__ dst,
                                int rows, int cols) {
  int idx = blockIdx.x * blockDim.x + threadIdx.x;
  if (idx >= rows * cols) return;
  int r = idx / cols, c = idx % cols;
  dst[(size_t)c * rows + r] = (_Float16)src[idx];
}

// ---------- kernel 2: QKV projection GEMM, 2M x 4N register blocking per wave ----------
// macro-tiles: M 32-rows x N 64-cols ; per K-step: 2 A-loads + 4 B-loads -> 8 WMMAs
__global__ __launch_bounds__(256) void k_qkv_gemm(
    const float* __restrict__ X, const _Float16* __restrict__ Wt,
    const float* __restrict__ bqkv,
    _Float16* __restrict__ Q, _Float16* __restrict__ K, _Float16* __restrict__ Vt) {
  const int NMT = QKVC / 64;  // 18 macro column tiles
  int wave = blockIdx.x * (blockDim.x >> 5) + (threadIdx.x >> 5);
  int mt = wave / NMT, nt = wave % NMT;   // mt: 32-row block, nt: 64-col block
  int row0 = mt * 32, col0 = nt * 64;

  v8f acc[2][4];
#pragma unroll
  for (int i = 0; i < 2; ++i)
#pragma unroll
    for (int j = 0; j < 4; ++j) acc[i][j] = (v8f){};

  for (int ks = 0; ks < DIM_; ks += 32) {
    v16h a0 = load_a_f32cvt(X, DIM_, row0,      ks);
    v16h a1 = load_a_f32cvt(X, DIM_, row0 + 16, ks);
#pragma unroll
    for (int j = 0; j < 4; ++j) {
      v16h b = load_b_kcont(Wt, DIM_, col0 + j * 16, ks);
      acc[0][j] = WMMA_F16(a0, b, acc[0][j]);
      acc[1][j] = WMMA_F16(a1, b, acc[1][j]);
    }
  }

#pragma unroll
  for (int j = 0; j < 4; ++j) {
    int n = col0 + j * 16 + D_COL();
    int three = n / DIM_;            // 0=q 1=k 2=v (uniform per 16-col subtile)
    int h = (n % DIM_) / HD_;        // uniform per subtile
    int d = n % HD_;
    float bias = bqkv[n];
    float scl = (three == 0) ? 0.17677669529663687f : 1.0f;  // HD^-0.5
#pragma unroll
    for (int i = 0; i < 2; ++i) {
#pragma unroll
      for (int e = 0; e < 8; ++e) {
        int r = row0 + i * 16 + D_ROW(e);
        int win = r / NTOK, tok = r % NTOK;
        _Float16 val = (_Float16)((acc[i][j][e] + bias) * scl);
        int bh = win * HEADS_ + h;
        if (three == 0)      Q[((size_t)bh * NTOK + tok) * HD_ + d] = val;
        else if (three == 1) K[((size_t)bh * NTOK + tok) * HD_ + d] = val;
        else                 Vt[((size_t)bh * HD_ + d) * NTOK + tok] = val;
      }
    }
  }
}

// ---------- kernel 3: attention (one wave per (win,head,16-row block)) ----------
__global__ __launch_bounds__(32) void k_attn(
    const _Float16* __restrict__ Q, const _Float16* __restrict__ K,
    const _Float16* __restrict__ Vt,
    const float* __restrict__ bias_table, const int* __restrict__ pos_idx,
    const float* __restrict__ mask, _Float16* __restrict__ AO) {
  __shared__ float    S[16][NTOK];
  __shared__ _Float16 P[16][160];
  __shared__ float    red[32];

  int wg = blockIdx.x;
  int rb = wg % 9;
  int bh = wg / 9;
  int h = bh % HEADS_, b = bh / HEADS_;
  int lane = threadIdx.x;
  const _Float16* Qb = Q + (size_t)bh * NTOK * HD_;
  const _Float16* Kb = K + (size_t)bh * NTOK * HD_;
  const _Float16* Vb = Vt + (size_t)bh * HD_ * NTOK;
  const float* maskb = mask + (size_t)(b & 7) * NTOK * NTOK;
  const float* btab  = bias_table + (size_t)(b & 31) * HEADS_ + h;  // + tb*32*12
  int brep = b >> 5;
  int i0 = rb * 16;

  v16h aq = load_a_f16(Qb, HD_, i0, 0);   // Q block, K-dim = HD = 32 exactly

  for (int t = 0; t < 9; ++t) {
    v16h bk = load_b_kcont(Kb, HD_, t * 16, 0);
    v8f s = {};
    s = WMMA_F16(aq, bk, s);
    int j = t * 16 + (lane & 15);
#pragma unroll
    for (int e = 0; e < 8; ++e) {
      int lr = D_ROW(e);
      int i = i0 + lr;
      int q = i * NTOK + j;
      // reference's tiled-gather algebra: table[idx[(q%2592)*8 + b/32], b%32, h]
      int pos = (q % 2592) * 8 + brep;
      int tb = pos_idx[pos];
      float bias = btab[(size_t)tb * (32 * HEADS_)];
      S[lr][j] = s[e] + bias + maskb[q];
    }
  }
  __syncthreads();

  // softmax: lane -> (row = lane&15, half = lane>>4), 72 cols per half
  {
    int r = lane & 15, half = lane >> 4;
    int c0 = half * 72;
    float mx = -3.0e38f;
    for (int c = c0; c < c0 + 72; ++c) mx = fmaxf(mx, S[r][c]);
    red[lane] = mx;
    __syncthreads();
    mx = fmaxf(mx, red[lane ^ 16]);
    float sum = 0.f;
    for (int c = c0; c < c0 + 72; ++c) { float e = __expf(S[r][c] - mx); S[r][c] = e; sum += e; }
    red[lane] = sum;
    __syncthreads();
    sum += red[lane ^ 16];
    float inv = 1.0f / sum;
    for (int c = c0; c < c0 + 72; ++c) P[r][c] = (_Float16)(S[r][c] * inv);
    for (int c = NTOK + half * 8; c < NTOK + half * 8 + 8; ++c) P[r][c] = (_Float16)0.f;  // zero pad to 160
  }
  __syncthreads();

  // PV: out block 16x32, K-dim = 144 padded to 160 -> 5 chunks of 32
#pragma unroll
  for (int ntile = 0; ntile < 2; ++ntile) {
    v8f acc = {};
    for (int mc = 0; mc < 5; ++mc) {
      v16h ap;
      {
        int m = lane & 15;
        int kb = mc * 32 + ((lane >> 4) << 3);
#pragma unroll
        for (int e = 0; e < 8; ++e) { ap[e] = P[m][kb + e]; ap[e + 8] = P[m][kb + 16 + e]; }
      }
      v16h bv;
      {
        int d = ntile * 16 + (lane & 15);
        int kb = mc * 32 + ((lane >> 4) << 3);
#pragma unroll
        for (int e = 0; e < 16; ++e) {
          int tok = kb + (e < 8 ? e : e + 8);
          tok = tok > (NTOK - 1) ? (NTOK - 1) : tok;  // clamp; P pad is 0 so contribution is 0
          bv[e] = Vb[(size_t)d * NTOK + tok];
        }
      }
      acc = WMMA_F16(ap, bv, acc);
    }
#pragma unroll
    for (int e = 0; e < 8; ++e) {
      int tok = i0 + D_ROW(e);
      int dd = ntile * 16 + D_COL();
      AO[((size_t)b * NTOK + tok) * DIM_ + h * HD_ + dd] = (_Float16)acc[e];
    }
  }
}

// ---------- kernel 4: output projection GEMM, 2M x 4N register blocking ----------
__global__ __launch_bounds__(256) void k_out_gemm(
    const _Float16* __restrict__ AO, const _Float16* __restrict__ WoT,
    const float* __restrict__ bout, float* __restrict__ out) {
  const int NMT = DIM_ / 64;  // 6 macro column tiles
  int wave = blockIdx.x * (blockDim.x >> 5) + (threadIdx.x >> 5);
  int mt = wave / NMT, nt = wave % NMT;
  int row0 = mt * 32, col0 = nt * 64;

  v8f acc[2][4];
#pragma unroll
  for (int i = 0; i < 2; ++i)
#pragma unroll
    for (int j = 0; j < 4; ++j) acc[i][j] = (v8f){};

  for (int ks = 0; ks < DIM_; ks += 32) {
    v16h a0 = load_a_f16(AO, DIM_, row0,      ks);
    v16h a1 = load_a_f16(AO, DIM_, row0 + 16, ks);
#pragma unroll
    for (int j = 0; j < 4; ++j) {
      v16h b = load_b_kcont(WoT, DIM_, col0 + j * 16, ks);
      acc[0][j] = WMMA_F16(a0, b, acc[0][j]);
      acc[1][j] = WMMA_F16(a1, b, acc[1][j]);
    }
  }

#pragma unroll
  for (int j = 0; j < 4; ++j) {
    int c = col0 + j * 16 + D_COL();
    float bias = bout[c];
#pragma unroll
    for (int i = 0; i < 2; ++i) {
#pragma unroll
      for (int e = 0; e < 8; ++e) {
        int r = row0 + i * 16 + D_ROW(e);
        out[(size_t)r * DIM_ + c] = acc[i][j][e] + bias;
      }
    }
  }
}

extern "C" void kernel_launch(void* const* d_in, const int* in_sizes, int n_in,
                              void* d_out, int out_size, void* d_ws, size_t ws_size,
                              hipStream_t stream) {
  (void)in_sizes; (void)n_in; (void)out_size; (void)ws_size;
  const float* x          = (const float*)d_in[0];
  const float* mask       = (const float*)d_in[1];
  const float* w_qkv      = (const float*)d_in[2];
  const float* b_qkv      = (const float*)d_in[3];
  const float* w_out      = (const float*)d_in[4];
  const float* b_out      = (const float*)d_in[5];
  const float* bias_table = (const float*)d_in[6];
  const int*   pos_idx    = (const int*)d_in[7];
  float* out = (float*)d_out;

  char* ws = (char*)d_ws;
  _Float16* Wt  = (_Float16*)(ws);                       // 1152x384 f16      (884736 B)
  _Float16* WoT = (_Float16*)(ws + 884736);              // 384x384 f16       (294912 B)
  _Float16* Q   = (_Float16*)(ws + 1179648);             // [bh][tok][d]      (28311552 B)
  _Float16* K   = (_Float16*)(ws + 29491200);            // [bh][tok][d]      (28311552 B)
  _Float16* Vt  = (_Float16*)(ws + 57802752);            // [bh][d][tok]      (28311552 B)
  _Float16* AO  = (_Float16*)(ws + 86114304);            // [row][384] f16    (28311552 B)

  k_transpose_f16<<<(DIM_ * QKVC + 255) / 256, 256, 0, stream>>>(w_qkv, Wt, DIM_, QKVC);
  k_transpose_f16<<<(DIM_ * DIM_ + 255) / 256, 256, 0, stream>>>(w_out, WoT, DIM_, DIM_);

  // (36864/32) x (1152/64) = 1152 x 18 = 20736 macro-tiles, 8 waves per 256-thread block
  k_qkv_gemm<<<(ROWS_ / 32) * (QKVC / 64) / 8, 256, 0, stream>>>(x, Wt, b_qkv, Q, K, Vt);

  // 256 windows * 12 heads * 9 row-blocks = 27648 single-wave workgroups
  k_attn<<<BW_ * HEADS_ * 9, 32, 0, stream>>>(Q, K, Vt, bias_table, pos_idx, mask, AO);

  // (36864/32) x (384/64) = 1152 x 6 = 6912 macro-tiles, 8 per block
  k_out_gemm<<<(ROWS_ / 32) * (DIM_ / 64) / 8, 256, 0, stream>>>(AO, WoT, b_out, out);
}
